// MultiHeadDifferentialAttention_2808908611956
// MI455X (gfx1250) — compile-verified, matrix operand order checked by reading, not run on hardware
//
#include <hip/hip_runtime.h>
#include <hip/hip_bf16.h>

#define B_ 2
#define S_ 2048
#define D_ 1024
#define H_ 16
#define DH_ 64
#define NT (B_*S_)          // 4096 tokens
#define GN_EPS 1e-5f
#define LAMBDA_INIT 0.8f

typedef __attribute__((ext_vector_type(16))) __bf16 v16bf;
typedef __attribute__((ext_vector_type(8)))  float  v8f;

struct __align__(16) U128 { unsigned int x[4]; };
union Frag { v16bf v; U128 q[2]; };

#if defined(__has_builtin)
#if __has_builtin(__builtin_amdgcn_tensor_load_to_lds) && __has_builtin(__builtin_amdgcn_s_wait_tensorcnt)
#define HAVE_TDM 1
#endif
#endif

__device__ __forceinline__ unsigned short f2bf(float f) {
    unsigned int u = __float_as_uint(f);
    u += 0x7fffu + ((u >> 16) & 1u);           // round-to-nearest-even
    return (unsigned short)(u >> 16);
}

// ---- Tensor Data Mover: async 2D bf16 tile (tile_w x tile_h) global -> LDS ----
// Builds D# group0/group1 per CDNA5 ISA: count=1, type=2 (image),
// data_size=2B, tile_dim0=tile_w (contiguous), tile_dim1=tile_h rows,
// tensor_dim0_stride=stride_elems. Issued once per calling wave (EXEC ignored).
__device__ __forceinline__ void tdm_load_2d(unsigned short* lptr, const void* gptr,
                                            unsigned tile_w, unsigned tile_h,
                                            unsigned td0, unsigned td1,
                                            unsigned stride_elems)
{
#ifdef HAVE_TDM
    typedef unsigned int v4u __attribute__((ext_vector_type(4)));
    typedef int v8i __attribute__((ext_vector_type(8)));
    typedef int v4i __attribute__((ext_vector_type(4)));
    unsigned long long ga = (unsigned long long)gptr;
    unsigned lds_off = (unsigned)(unsigned long long)lptr;   // LDS aperture: low bits
    v4u g0;
    g0[0] = 1u;                                   // count=1, is_restore=0, gather=0
    g0[1] = lds_off;                              // lds_addr (bits 63:32)
    g0[2] = (unsigned)ga;                         // global_addr[31:0]  (bits 95:64)
    g0[3] = (unsigned)(ga >> 32) | (2u << 30);    // global_addr[56:32] | type=2
    v8i g1;
    g1[0] = 0x00010000;                           // wg_mask=0, data_size=1 (2 bytes)
    g1[1] = (int)((td0 & 0xffffu) << 16);         // tensor_dim0 low (bits 79:48)
    g1[2] = (int)((td0 >> 16) | ((td1 & 0xffffu) << 16)); // dim0 hi | dim1 lo
    g1[3] = (int)((td1 >> 16) | (tile_w << 16));  // dim1 hi | tile_dim0
    g1[4] = (int)tile_h;                          // tile_dim1 (tile_dim2=0)
    g1[5] = (int)stride_elems;                    // tensor_dim0_stride low 32
    g1[6] = 0;                                    // stride hi | dim1_stride lo
    g1[7] = 0;
    v4i z4 = {0, 0, 0, 0};
#if __clang_major__ >= 23
    v8i z8 = {0, 0, 0, 0, 0, 0, 0, 0};
    __builtin_amdgcn_tensor_load_to_lds(g0, g1, z4, z4, z8, 0);
#else
    __builtin_amdgcn_tensor_load_to_lds(g0, g1, z4, z4, 0);
#endif
#else
    // Fallback: cooperative copy by the calling wave (32 lanes)
    const unsigned short* g = (const unsigned short*)gptr;
    int lane = threadIdx.x & 31;
    int total = (int)(tile_w * tile_h) / 8;
    for (int idx = lane; idx < total; idx += 32) {
        int row = (idx * 8) / (int)tile_w;
        int col = (idx * 8) % (int)tile_w;
        *(U128*)(lptr + row * tile_w + col) =
            *(const U128*)(g + (size_t)row * stride_elems + col);
    }
    (void)td0; (void)td1;
#endif
}

template <int N>
__device__ __forceinline__ void tdm_wait() {
#ifdef HAVE_TDM
    __builtin_amdgcn_s_wait_tensorcnt(N);      // SIMM16: compile-time constant
#endif
}

// ---------------- elementwise convert fp32 -> bf16 ----------------
__global__ void cvt_f32_to_bf16(const float* __restrict__ in,
                                unsigned short* __restrict__ out, int n) {
    int i = blockIdx.x * blockDim.x + threadIdx.x;
    if (i < n) out[i] = f2bf(in[i]);
}

// ---- convert + transpose 1024x1024 weight: Wt[n][k] = W[k][n] ----
__global__ void cvt_transpose_1024(const float* __restrict__ W,
                                   unsigned short* __restrict__ Wt) {
    int i = blockIdx.x * blockDim.x + threadIdx.x;   // i = n*1024 + k
    int n = i >> 10, k = i & 1023;
    Wt[i] = f2bf(W[k * 1024 + n]);
}

__global__ void zero_f32(float* p, int n) {
    int i = blockIdx.x * blockDim.x + threadIdx.x;
    if (i < n) p[i] = 0.f;
}

// ---------------- per-head differential lambda ----------------
__global__ void lambda_kernel(const float* __restrict__ lq1, const float* __restrict__ lk1,
                              const float* __restrict__ lq2, const float* __restrict__ lk2,
                              float* __restrict__ lam) {
    int h = threadIdx.x;
    if (h < H_) {
        float d1 = 0.f, d2 = 0.f;
        for (int i = 0; i < DH_; ++i) {
            d1 += lq1[h * DH_ + i] * lk1[h * DH_ + i];
            d2 += lq2[h * DH_ + i] * lk2[h * DH_ + i];
        }
        lam[h] = __expf(d1) - __expf(d2) + LAMBDA_INIT;
    }
}

// ---- WMMA GEMM: C[M,N] = A[M,K](bf16) @ Bt[N,K]^T(bf16) + bias ----
// 8 waves/block; wave w -> 16 rows, block -> 128 rows x 64 cols.
// B tile (64 cols x 32 k) staged once per block via TDM, double-buffered.
__global__ __launch_bounds__(256) void gemm_bf16_wmma(
    const unsigned short* __restrict__ A,
    const unsigned short* __restrict__ Bt,
    const float* __restrict__ bias,
    void* __restrict__ C, int M, int N, int K, int out_bf16)
{
    __shared__ __align__(16) unsigned short Bs[2][64][32];   // 2 x 4 KB

    const int lane  = threadIdx.x & 31;
    const int w     = threadIdx.x >> 5;
    const int M0    = blockIdx.y * 128 + w * 16;
    const int N0    = blockIdx.x * 64;
    const int cn    = lane & 15;
    const int khalf = (lane >> 4) * 8;

    v8f zf = {0.f,0.f,0.f,0.f,0.f,0.f,0.f,0.f};
    v8f acc[4];
    #pragma unroll
    for (int n = 0; n < 4; ++n) acc[n] = zf;

    const unsigned short* pa_base = A + (size_t)(M0 + cn) * K + khalf;
    const unsigned short* pb_base = Bt + (size_t)N0 * K;

    if (w == 0)
        tdm_load_2d(&Bs[0][0][0], pb_base, 32, 64, (unsigned)K, (unsigned)N, (unsigned)K);

    int buf = 0;
    for (int k0 = 0; k0 < K; k0 += 32) {
        if (w == 0) {
            if (k0 + 32 < K) {
                tdm_load_2d(&Bs[buf ^ 1][0][0], pb_base + k0 + 32, 32, 64,
                            (unsigned)K, (unsigned)N, (unsigned)K);
                tdm_wait<1>();     // previous tile complete (in-order TDM)
            } else {
                tdm_wait<0>();     // last tile: drain
            }
        }
        __syncthreads();

        Frag a;
        a.q[0] = *(const U128*)(pa_base + k0);
        a.q[1] = *(const U128*)(pa_base + k0 + 16);
        __builtin_prefetch(pa_base + k0 + 64, 0, 1);   // global_prefetch_b8
        #pragma unroll
        for (int n = 0; n < 4; ++n) {
            const unsigned short* pb = &Bs[buf][n * 16 + cn][khalf];
            Frag bf;
            bf.q[0] = *(const U128*)(pb);
            bf.q[1] = *(const U128*)(pb + 16);
            acc[n] = __builtin_amdgcn_wmma_f32_16x16x32_bf16(
                false, a.v, false, bf.v, (short)0, acc[n], false, false);
        }
        __syncthreads();
        buf ^= 1;
    }

    const int rhalf = (lane >> 4) * 8;
    #pragma unroll
    for (int n = 0; n < 4; ++n) {
        int col = N0 + n * 16 + cn;
        float bv = bias ? bias[col] : 0.f;
        #pragma unroll
        for (int r = 0; r < 8; ++r) {
            int row = M0 + r + rhalf;
            float val = acc[n][r] + bv;
            if (out_bf16) ((unsigned short*)C)[(size_t)row * N + col] = f2bf(val);
            else          ((float*)C)[(size_t)row * N + col] = val;
        }
    }
}

// ---- transpose V: Vt[b,h,d,j] = V[b,j,h,d] (bf16) ----
__global__ void transpose_v(const unsigned short* __restrict__ Vb,
                            unsigned short* __restrict__ Vt) {
    int i = blockIdx.x * blockDim.x + threadIdx.x;   // over B*S*D
    int j = i & (S_ - 1);
    int t = i >> 11;                                  // (b*H+h)*DH + d
    int d = t & (DH_ - 1);
    int h = (t >> 6) & (H_ - 1);
    int b = t >> 10;
    Vt[i] = Vb[(size_t)b * S_ * D_ + (size_t)j * D_ + h * DH_ + d];
}

// ---------------- fused flash differential attention ----------------
// grid = (S/128, B*H); block 256 (8 waves); wave = 16 query rows, full DH=64.
// K and V 64x64 tiles are shared by all 8 waves -> staged via TDM, double-buffered.
__global__ __launch_bounds__(256) void diff_attn_kernel(
    const unsigned short* __restrict__ Q,    // [B,S,H,DH] bf16
    const unsigned short* __restrict__ Kd,   // [B,S,H,DH] bf16
    const unsigned short* __restrict__ Vt,   // [B,H,DH,S] bf16
    const float* __restrict__ lam,
    float* __restrict__ O,                   // [B,S,H,DH] f32
    float* __restrict__ gnsum)               // [B*H][2] (sum, sumsq)
{
    __shared__ __align__(16) unsigned short Ks[2][64][64];   // 2 x 8 KB
    __shared__ __align__(16) unsigned short Vs[2][64][64];   // 2 x 8 KB
    __shared__ __align__(16) unsigned short Pbuf[8][16][64]; // 16 KB

    const int lane  = threadIdx.x & 31;
    const int w     = threadIdx.x >> 5;
    const int bh    = blockIdx.y;
    const int b     = bh >> 4, h = bh & 15;
    const int i0    = blockIdx.x * 128 + w * 16;
    const int cn    = lane & 15;
    const int khalf = (lane >> 4) * 8;

    const float sl = 0.125f * lam[h];        // DH^-0.5 * lambda

    // Q fragments for this wave's 16 rows (K = DH = 64 -> two 32-wide steps)
    const size_t qbase = (size_t)b * S_ * D_ + (size_t)(i0 + cn) * D_ + h * DH_ + khalf;
    Frag qf[2];
    qf[0].q[0] = *(const U128*)(Q + qbase);
    qf[0].q[1] = *(const U128*)(Q + qbase + 16);
    qf[1].q[0] = *(const U128*)(Q + qbase + 32);
    qf[1].q[1] = *(const U128*)(Q + qbase + 48);

    v8f zf = {0.f,0.f,0.f,0.f,0.f,0.f,0.f,0.f};
    v8f acc[4];
    #pragma unroll
    for (int n = 0; n < 4; ++n) acc[n] = zf;
    float m[8], l[8];
    #pragma unroll
    for (int r = 0; r < 8; ++r) { m[r] = -3.0e38f; l[r] = 0.f; }

    const unsigned short* kbase = Kd + (size_t)b * S_ * D_ + h * DH_;   // row j stride D
    const unsigned short* vbase = Vt + (size_t)(b * H_ + h) * DH_ * S_; // row d stride S

    if (w == 0) {
        tdm_load_2d(&Ks[0][0][0], kbase, 64, 64, (unsigned)D_, (unsigned)S_, (unsigned)D_);
        tdm_load_2d(&Vs[0][0][0], vbase, 64, 64, (unsigned)S_, (unsigned)DH_, (unsigned)S_);
    }

    int buf = 0;
    for (int jt = 0; jt < S_; jt += 64) {
        if (w == 0) {
            if (jt + 64 < S_) {
                tdm_load_2d(&Ks[buf ^ 1][0][0], kbase + (size_t)(jt + 64) * D_, 64, 64,
                            (unsigned)D_, (unsigned)S_, (unsigned)D_);
                tdm_load_2d(&Vs[buf ^ 1][0][0], vbase + jt + 64, 64, 64,
                            (unsigned)S_, (unsigned)DH_, (unsigned)S_);
                tdm_wait<2>();     // current pair complete, next pair in flight
            } else {
                tdm_wait<0>();     // last tile pair: drain
            }
        }
        __syncthreads();

        // ---- scores: S16x64 = Q16x64 @ K^T (K tile from LDS) ----
        v8f s[4];
        #pragma unroll
        for (int n = 0; n < 4; ++n) s[n] = zf;
        #pragma unroll
        for (int kk = 0; kk < 2; ++kk) {
            #pragma unroll
            for (int n = 0; n < 4; ++n) {
                const unsigned short* pk = &Ks[buf][n * 16 + cn][kk * 32 + khalf];
                Frag kf;
                kf.q[0] = *(const U128*)pk;
                kf.q[1] = *(const U128*)(pk + 16);
                s[n] = __builtin_amdgcn_wmma_f32_16x16x32_bf16(
                    false, qf[kk].v, false, kf.v, (short)0, s[n], false, false);
            }
        }
        // ---- online softmax (rows live across 16-lane half-waves) ----
        float rmax[8];
        #pragma unroll
        for (int r = 0; r < 8; ++r) {
            float v = -3.0e38f;
            #pragma unroll
            for (int n = 0; n < 4; ++n) { s[n][r] *= sl; v = fmaxf(v, s[n][r]); }
            #pragma unroll
            for (int off = 1; off < 16; off <<= 1) v = fmaxf(v, __shfl_xor(v, off, 16));
            rmax[r] = v;
        }
        float corr[8];
        #pragma unroll
        for (int r = 0; r < 8; ++r) {
            float mn = fmaxf(m[r], rmax[r]);
            corr[r] = __expf(m[r] - mn);
            m[r] = mn;
        }
        float psum[8];
        #pragma unroll
        for (int r = 0; r < 8; ++r) psum[r] = 0.f;
        #pragma unroll
        for (int n = 0; n < 4; ++n) {
            #pragma unroll
            for (int r = 0; r < 8; ++r) {
                float p = __expf(s[n][r] - m[r]);
                psum[r] += p;
                acc[n][r] *= corr[r];
                Pbuf[w][r + khalf][n * 16 + cn] = f2bf(p);
            }
        }
        #pragma unroll
        for (int r = 0; r < 8; ++r) {
            float ps = psum[r];
            #pragma unroll
            for (int off = 1; off < 16; off <<= 1) ps += __shfl_xor(ps, off, 16);
            l[r] = l[r] * corr[r] + ps;
        }
        __syncthreads();
        // ---- O16x64 += P16x64 @ V64x64 (P and V tiles from LDS) ----
        #pragma unroll
        for (int kk = 0; kk < 2; ++kk) {
            const unsigned short* pp = &Pbuf[w][cn][kk * 32 + khalf];
            Frag pf;
            pf.q[0] = *(const U128*)pp;
            pf.q[1] = *(const U128*)(pp + 16);
            #pragma unroll
            for (int n = 0; n < 4; ++n) {
                const unsigned short* pv = &Vs[buf][n * 16 + cn][kk * 32 + khalf];
                Frag vf;
                vf.q[0] = *(const U128*)pv;
                vf.q[1] = *(const U128*)(pv + 16);
                acc[n] = __builtin_amdgcn_wmma_f32_16x16x32_bf16(
                    false, pf.v, false, vf.v, (short)0, acc[n], false, false);
            }
        }
        __syncthreads();
        buf ^= 1;
    }

    // ---- finalize: divide by l, store fp32, accumulate GroupNorm stats ----
    float inv[8];
    #pragma unroll
    for (int r = 0; r < 8; ++r) inv[r] = 1.f / l[r];
    float s1 = 0.f, s2 = 0.f;
    #pragma unroll
    for (int n = 0; n < 4; ++n) {
        #pragma unroll
        for (int r = 0; r < 8; ++r) {
            float v = acc[n][r] * inv[r];
            size_t row = (size_t)(i0 + r + khalf);
            O[(size_t)b * S_ * D_ + row * D_ + h * DH_ + n * 16 + cn] = v;
            s1 += v; s2 += v * v;
        }
    }
    #pragma unroll
    for (int off = 1; off < 32; off <<= 1) {
        s1 += __shfl_xor(s1, off, 32);
        s2 += __shfl_xor(s2, off, 32);
    }
    if (lane == 0) {
        atomicAdd(&gnsum[bh * 2 + 0], s1);
        atomicAdd(&gnsum[bh * 2 + 1], s2);
    }
}

// ---------------- GroupNorm normalize + affine -> bf16 ----------------
__global__ void gn_kernel(const float* __restrict__ O, const float* __restrict__ gnsum,
                          const float* __restrict__ gw, const float* __restrict__ gb,
                          unsigned short* __restrict__ normed, int n) {
    int i = blockIdx.x * blockDim.x + threadIdx.x;
    if (i >= n) return;
    int c = i & (D_ - 1);
    int h = c >> 6;
    int b = i / (S_ * D_);
    float sum = gnsum[(b * H_ + h) * 2 + 0];
    float sq  = gnsum[(b * H_ + h) * 2 + 1];
    const float invN = 1.f / (float)(S_ * DH_);
    float mean = sum * invN;
    float var  = sq * invN - mean * mean;
    float rs   = rsqrtf(var + GN_EPS);
    normed[i] = f2bf((O[i] - mean) * rs * gw[c] + gb[c]);
}

// =====================================================================
extern "C" void kernel_launch(void* const* d_in, const int* in_sizes, int n_in,
                              void* d_out, int out_size, void* d_ws, size_t ws_size,
                              hipStream_t stream) {
    const float* x   = (const float*)d_in[0];
    const float* Wq  = (const float*)d_in[1];
    const float* bq  = (const float*)d_in[2];
    const float* Wk  = (const float*)d_in[3];
    const float* bk  = (const float*)d_in[4];
    const float* Wv  = (const float*)d_in[5];
    const float* bv  = (const float*)d_in[6];
    const float* Wo  = (const float*)d_in[7];
    const float* bo  = (const float*)d_in[8];
    const float* lq1 = (const float*)d_in[9];
    const float* lk1 = (const float*)d_in[10];
    const float* lq2 = (const float*)d_in[11];
    const float* lk2 = (const float*)d_in[12];
    const float* gw  = (const float*)d_in[13];
    const float* gb  = (const float*)d_in[14];

    char* ws = (char*)d_ws;
    size_t off = 0;
    auto alloc = [&](size_t bytes) -> void* {
        void* p = ws + off;
        off += (bytes + 255) & ~(size_t)255;
        return p;
    };
    const size_t tokBytesBf = (size_t)NT * D_ * 2;   // 8 MB
    const size_t wBytesBf   = (size_t)D_ * D_ * 2;   // 2 MB

    unsigned short* xb   = (unsigned short*)alloc(tokBytesBf);
    unsigned short* WqT  = (unsigned short*)alloc(wBytesBf);
    unsigned short* WkT  = (unsigned short*)alloc(wBytesBf);
    unsigned short* WvT  = (unsigned short*)alloc(wBytesBf);
    unsigned short* WoT  = (unsigned short*)alloc(wBytesBf);
    unsigned short* Qb   = (unsigned short*)alloc(tokBytesBf);
    unsigned short* Kb   = (unsigned short*)alloc(tokBytesBf);
    unsigned short* Vb   = (unsigned short*)alloc(tokBytesBf);
    unsigned short* Vt   = (unsigned short*)alloc(tokBytesBf);
    float*          Obuf = (float*)alloc((size_t)NT * D_ * 4);
    unsigned short* Nrm  = (unsigned short*)alloc(tokBytesBf);
    float*          lamb = (float*)alloc(256);
    float*          gns  = (float*)alloc(256);

    const int nTok = NT * D_;                 // 4,194,304
    const int nW   = D_ * D_;                 // 1,048,576

    cvt_f32_to_bf16<<<nTok / 256, 256, 0, stream>>>(x, xb, nTok);
    cvt_transpose_1024<<<nW / 256, 256, 0, stream>>>(Wq, WqT);
    cvt_transpose_1024<<<nW / 256, 256, 0, stream>>>(Wk, WkT);
    cvt_transpose_1024<<<nW / 256, 256, 0, stream>>>(Wv, WvT);
    cvt_transpose_1024<<<nW / 256, 256, 0, stream>>>(Wo, WoT);
    lambda_kernel<<<1, 32, 0, stream>>>(lq1, lk1, lq2, lk2, lamb);
    zero_f32<<<1, 64, 0, stream>>>(gns, B_ * H_ * 2);

    dim3 ggrid(D_ / 64, NT / 128);            // (16, 32)
    gemm_bf16_wmma<<<ggrid, 256, 0, stream>>>(xb, WqT, bq, Qb, NT, D_, D_, 1);
    gemm_bf16_wmma<<<ggrid, 256, 0, stream>>>(xb, WkT, bk, Kb, NT, D_, D_, 1);
    gemm_bf16_wmma<<<ggrid, 256, 0, stream>>>(xb, WvT, bv, Vb, NT, D_, D_, 1);

    transpose_v<<<nTok / 256, 256, 0, stream>>>(Vb, Vt);

    dim3 agrid(S_ / 128, B_ * H_);            // (16, 32)
    diff_attn_kernel<<<agrid, 256, 0, stream>>>(Qb, Kb, Vt, lamb, Obuf, gns);

    gn_kernel<<<nTok / 256, 256, 0, stream>>>(Obuf, gns, gw, gb, Nrm, nTok);

    gemm_bf16_wmma<<<ggrid, 256, 0, stream>>>(Nrm, WoT, bo, (float*)d_out, NT, D_, D_, 0);
}
